// OrientationMap_17274358464646
// MI455X (gfx1250) — matrix operand order repaired and checked by reading, not compiled, not run on hardware
//
#include <hip/hip_runtime.h>

// ---------------------------------------------------------------------------
// OrientationMap on MI455X (gfx1250, wave32).
// Output = 268.6 MB (mostly a zero tensor) -> HBM-store-bound (~11.5us floor
// at 23.3 TB/s). Compute (~0.5 GFLOP of 15x64x64-ish matmuls) is negligible,
// done with v_wmma_f32_16x16x32_f16 (f32 accumulation).
// ---------------------------------------------------------------------------

typedef __attribute__((ext_vector_type(16))) _Float16 v16h;
typedef __attribute__((ext_vector_type(8)))  float    v8f;
typedef __attribute__((ext_vector_type(4)))  float    v4f;

#define HH   64
#define WW   64
#define FS   15
#define RAD  7
#define BATCH 4
#define PIX  (HH*WW)                 // 4096
#define RESP_ELEMS ((size_t)BATCH * PIX * PIX)   // 67,108,864 floats

// ---------------------------------------------------------------------------
// Kernel 1: zero-fill the responses tensor with wide non-temporal stores.
// 16.7M float4 stores; grid-stride so every lane issues back-to-back b128 NT
// stores -> saturates HBM write bandwidth.
// ---------------------------------------------------------------------------
__global__ __launch_bounds__(256) void fill_zero_kernel(v4f* __restrict__ p,
                                                        long long n4) {
    long long idx    = (long long)blockIdx.x * blockDim.x + threadIdx.x;
    long long stride = (long long)gridDim.x * blockDim.x;
    v4f z = {0.0f, 0.0f, 0.0f, 0.0f};
    for (; idx < n4; idx += stride) {
        __builtin_nontemporal_store(z, p + idx);
    }
}

// Resize-adjoint element: value of R_p[o, M], R_p = bilinear resize matrix
// [p, 15] (align_corners=False), exactly matching the reference _resize_mat.
// For p == 15 this degenerates to identity (interior pixels).
__device__ __forceinline__ float adjA(int M, int o, int p) {
    float c = fmaxf((o + 0.5f) * 15.0f / (float)p - 0.5f, 0.0f);
    int f = (int)floorf(c);
    if (f > FS - 1) f = FS - 1;
    float lam = c - (float)f;
    int f1 = (f + 1 > FS - 1) ? (FS - 1) : (f + 1);
    float v = 0.0f;
    if (M == f)  v += 1.0f - lam;
    if (M == f1) v += lam;
    return v;
}

// ---------------------------------------------------------------------------
// Kernel 2: one wave per (b, i) row pair; 256 waves total (32 blocks x 8 waves).
// Stage 1 (WMMA): t1(16x64) = AHpad_i(16x64) @ x_b(64x64), accumulated f32,
//   A-matrix synthesized analytically in-register (<=2 nonzeros per column),
//   4 N-tiles x 2 K-steps of v_wmma_f32_16x16x32_f16, result staged in LDS.
// Stage 2 (VALU): per output column j (2 per lane), fold the W-side adjoint
//   into the Gabor filter on the fly and reduce.
// ---------------------------------------------------------------------------
__global__ __launch_bounds__(256) void orientation_kernel(
    const float* __restrict__ x,      // [B,64,64]
    const float* __restrict__ gab,    // [4096,1,15,15]
    const float* __restrict__ field,  // [64,64]
    float* __restrict__ out)
{
    __shared__ float t1s[8][16][WW];          // 32 KB: one 16x64 slab per wave

    const int lane = threadIdx.x & 31;
    const int wv   = blockIdx.x * (blockDim.x >> 5) + (threadIdx.x >> 5);
    const int b    = wv >> 6;                 // 0..3
    const int i    = wv & 63;                 // 0..63

    int lo = i - RAD; if (lo < 0) lo = 0;
    int hi = i + RAD + 1; if (hi > HH) hi = HH;
    const int p = hi - lo;

    const int Mrow = lane & 15;               // A row / D column this lane owns
    const int half = lane >> 4;               // lane group 0/1
    const float* xb = x + b * PIX;

    float (*t1)[WW] = t1s[threadIdx.x >> 5];

    // ---- Stage 1: t1 = AHpad @ x_b via WMMA -------------------------------
    for (int n = 0; n < 4; ++n) {             // N-tiles of 16 columns
        v8f acc = {};
        for (int k = 0; k < 2; ++k) {         // K-tiles of 32
            // A fragment: ISA 16-bit A layout (16x32). Lane (M = lane&15,
            // half = lane>>4); VGPR r holds K = (r/4)*16 + half*8 + (r%4)*2.
            v16h a;
            for (int r = 0; r < 8; ++r) {
                int kk  = ((r >> 2) << 4) + (half << 3) + ((r & 3) << 1);
                int col = (k << 5) + kk;      // image row index h
                float a0 = 0.0f, a1 = 0.0f;
                if (Mrow < FS) {
                    int o0 = col - lo, o1 = col + 1 - lo;
                    if (o0 >= 0 && o0 < p) a0 = adjA(Mrow, o0, p);
                    if (o1 >= 0 && o1 < p) a1 = adjA(Mrow, o1, p);
                }
                a[2 * r]     = (_Float16)a0;
                a[2 * r + 1] = (_Float16)a1;
            }
            // B fragment: ISA 16-bit B layout (32x16). Lane owns column
            // N = lane&15; element h -> K = half*8 + (h&7) + (h>>3)*16.
            v16h bf;
            const int colw    = (n << 4) + (lane & 15);
            const int rowbase = (k << 5) + (half << 3);
            for (int h = 0; h < 16; ++h) {
                bf[h] = (_Float16)xb[(rowbase + (h & 7) + ((h >> 3) << 4)) * WW + colw];
            }
            acc = __builtin_amdgcn_wmma_f32_16x16x32_f16(
                      false, a, false, bf, (short)0, acc, false, false);
        }
        // D layout: lane column N = lane&15; VGPR r -> row M = r + 8*half.
        const int Ncol = (n << 4) + (lane & 15);
        for (int r = 0; r < 8; ++r) {
            t1[r + (half << 3)][Ncol] = acc[r];
        }
    }
    __syncthreads();

    // ---- Stage 2: fold W-adjoint + Gabor dot, write sparse outputs --------
    float* respOut = out;                                   // [B,4096,64,64]
    float* maxOut  = out + RESP_ELEMS;                      // [B,64,64]
    float* oriOut  = maxOut + (size_t)BATCH * PIX;          // [B,64,64]
    float* fldOut  = oriOut + (size_t)BATCH * PIX;          // [B,64,64]

    for (int jj = 0; jj < 2; ++jj) {
        const int j = lane + (jj << 5);
        int loj = j - RAD; if (loj < 0) loj = 0;
        int hij = j + RAD + 1; if (hij > WW) hij = WW;
        const int pj = hij - loj;
        const float* g = gab + (size_t)(i * WW + j) * (FS * FS);

        float resp = 0.0f;
        for (int o = 0; o < pj; ++o) {
            // Row o of R_pj has <=2 nonzeros at columns f (1-lam) and f1 (lam).
            float c = fmaxf((o + 0.5f) * 15.0f / (float)pj - 0.5f, 0.0f);
            int f = (int)floorf(c);
            if (f > FS - 1) f = FS - 1;
            float lam = c - (float)f;
            int f1 = (f + 1 > FS - 1) ? (FS - 1) : (f + 1);
            const int wcol = loj + o;
            for (int q = 0; q < FS; ++q) {    // q = filter row index p
                float t  = t1[q][wcol];
                float gv = (1.0f - lam) * g[q * FS + f] + lam * g[q * FS + f1];
                resp = fmaf(t, gv, resp);
            }
        }

        const int    k0   = i * WW + j;
        const size_t base = (size_t)b * PIX;
        // responses[b, k0, i, j]; flat = b*4096*4096 + k0*4096 + (i*64+j)=k0
        respOut[((size_t)b * PIX + (size_t)k0) * PIX + (size_t)k0] = resp;
        maxOut[base + k0] = fmaxf(resp, 0.0f);
        // argmax over axis 1 of the sparse tensor: only slot k0 holds resp,
        // all others are 0 -> jnp.argmax first-max semantics:
        int amax = (resp > 0.0f) ? k0 : ((resp < 0.0f && k0 == 0) ? 1 : 0);
        oriOut[base + k0] = field[amax];
        fldOut[base + k0] = field[k0];
    }
}

// ---------------------------------------------------------------------------
extern "C" void kernel_launch(void* const* d_in, const int* in_sizes, int n_in,
                              void* d_out, int out_size, void* d_ws, size_t ws_size,
                              hipStream_t stream) {
    (void)in_sizes; (void)n_in; (void)out_size; (void)d_ws; (void)ws_size;
    const float* x     = (const float*)d_in[0];   // [4,64,64]
    const float* gab   = (const float*)d_in[1];   // [4096,1,15,15]
    const float* field = (const float*)d_in[2];   // [64,64]
    float* out = (float*)d_out;

    // 1) Zero the 256 MB responses tensor (bandwidth-bound phase).
    const long long n4 = (long long)(RESP_ELEMS / 4);     // 16,777,216 float4
    fill_zero_kernel<<<4096, 256, 0, stream>>>((v4f*)out, n4);

    // 2) Compute responses diagonal + maps: 256 waves, WMMA stage-1.
    orientation_kernel<<<32, 256, 0, stream>>>(x, gab, field, out);
}
